// SequentialLSTM_20675972563344
// MI455X (gfx1250) — matrix operand order repaired and checked by reading, not compiled, not run on hardware
//
#include <hip/hip_runtime.h>
#include <math.h>

#define TT 512
#define BB 32
#define HH 512
#define G4 2048   // 4*HH

typedef __attribute__((ext_vector_type(16))) __bf16 v16bf;
typedef __attribute__((ext_vector_type(8)))  float  v8f;

union BV { uint4 u[2]; v16bf v; };

// A-matrix 16x32 bf16 fragment: per lane, 8 bf16 at kbase and 8 at kbase+16
// (lanes 0-15 hold K {0..7,16..23}, lanes 16-31 hold K {8..15,24..31};
//  caller passes kbase already offset by 8 for the upper half)
__device__ __forceinline__ v16bf ld_a_frag(const __bf16* row, int kbase) {
  BV r;
  r.u[0] = *(const uint4*)(row + kbase);
  r.u[1] = *(const uint4*)(row + kbase + 16);
  return r.v;
}

// B-matrix 32x16 bf16 fragment: per lane, 8 bf16 at p and 8 at p+16 (column of B =
// row of W; lanes 0-15 K {0..7,16..23}, lanes 16-31 K {8..15,24..31} — caller offsets)
__device__ __forceinline__ v16bf ld_b_frag(const __bf16* p) {
  BV r;
  r.u[0] = *(const uint4*)(p);
  r.u[1] = *(const uint4*)(p + 16);
  return r.v;
}

__device__ __forceinline__ v8f wmma_bf16(v16bf a, v16bf b, v8f c) {
  return __builtin_amdgcn_wmma_f32_16x16x32_bf16(
      /*neg_a=*/false, a, /*neg_b=*/false, b,
      /*c_mod=*/(short)0, c, /*reuse_a=*/false, /*reuse_b=*/false);
}

__device__ __forceinline__ float sigmoidf_(float x) {
  return 1.0f / (1.0f + __expf(-x));
}

// ---------------------------------------------------------------------------
// fp32 -> bf16 conversion
// ---------------------------------------------------------------------------
__global__ void f32_to_bf16_kernel(const float* __restrict__ src,
                                   __bf16* __restrict__ dst, int n) {
  int i = blockIdx.x * blockDim.x + threadIdx.x;
  if (i < n) dst[i] = (__bf16)src[i];
}

// ---------------------------------------------------------------------------
// Phase 1: xg[dir][b*T+t][g] = x[b][actual_t] . Wih_dir[g] + bih[g] + bhh[g]
// actual_t = t (fwd) or T-1-t (bwd). One WG = 64 rows x 256 cols, 8 waves,
// each wave = 4 M-tiles x 2 N-tiles of 16x16, K looped 512/32.
// ---------------------------------------------------------------------------
__global__ __launch_bounds__(256) void lstm_xgemm(
    const __bf16* __restrict__ xbf,        // [B*T, H] (natural time order)
    const __bf16* __restrict__ wih_f,      // [4H, H]
    const __bf16* __restrict__ wih_b,      // [4H, H]
    const float* __restrict__ bih_f, const float* __restrict__ bhh_f,
    const float* __restrict__ bih_b, const float* __restrict__ bhh_b,
    float* __restrict__ xg)                // [2, B*T, 4H]
{
  const int dir = blockIdx.z;
  const __bf16* W  = dir ? wih_b : wih_f;
  const float*  bi = dir ? bih_b : bih_f;
  const float*  bh = dir ? bhh_b : bhh_f;
  float* out = xg + (size_t)dir * (size_t)(BB * TT) * G4;

  const int wave  = threadIdx.x >> 5;
  const int lane  = threadIdx.x & 31;
  const int lm    = lane & 15;
  const int lhalf = lane >> 4;

  const int Mbase = blockIdx.x * 64;
  const int Nbase = blockIdx.y * 256 + wave * 32;

  // Per-lane A row pointers (4 M-tiles); bwd direction reads x reversed in t.
  const __bf16* arow[4];
#pragma unroll
  for (int mt = 0; mt < 4; ++mt) {
    int m = Mbase + mt * 16 + lm;
    int b = m >> 9;              // T == 512
    int ts = m & 511;
    int src_t = dir ? (TT - 1 - ts) : ts;
    arow[mt] = xbf + (size_t)(b * TT + src_t) * HH;
  }
  // Per-lane B row pointers (2 N-tiles): column n of B == row n of Wih.
  const __bf16* brow[2];
#pragma unroll
  for (int nt = 0; nt < 2; ++nt)
    brow[nt] = W + (size_t)(Nbase + nt * 16 + lm) * HH;

  v8f acc[4][2];
#pragma unroll
  for (int mt = 0; mt < 4; ++mt)
#pragma unroll
    for (int nt = 0; nt < 2; ++nt)
#pragma unroll
      for (int r = 0; r < 8; ++r) acc[mt][nt][r] = 0.0f;

  const int aoff = lhalf * 8;
  const int boff = lhalf * 8;
  for (int kb = 0; kb < HH; kb += 32) {
    v16bf bfrg[2];
#pragma unroll
    for (int nt = 0; nt < 2; ++nt)
      bfrg[nt] = ld_b_frag(brow[nt] + kb + boff);
#pragma unroll
    for (int mt = 0; mt < 4; ++mt) {
      v16bf afrg = ld_a_frag(arow[mt], kb + aoff);
#pragma unroll
      for (int nt = 0; nt < 2; ++nt)
        acc[mt][nt] = wmma_bf16(afrg, bfrg[nt], acc[mt][nt]);
    }
  }

  // Epilogue: C layout — lane holds column lm, rows lhalf*8 + r.
#pragma unroll
  for (int nt = 0; nt < 2; ++nt) {
    int col = Nbase + nt * 16 + lm;
    float bias = bi[col] + bh[col];
#pragma unroll
    for (int mt = 0; mt < 4; ++mt) {
      int rbase = Mbase + mt * 16 + lhalf * 8;
#pragma unroll
      for (int r = 0; r < 8; ++r)
        out[(size_t)(rbase + r) * G4 + col] = acc[mt][nt][r] + bias;
    }
  }
}

// ---------------------------------------------------------------------------
// Phase 2: persistent scan. 32 WGs: dir = wg>>4, slice of 32 hidden units.
// LDS: Whh slice [128 rows x 512] bf16 (128KB) + gate exchange [128][33] f32.
// Per step: gates = xg_t + h_{t-1} @ Whh_slice^T (WMMA), activations,
// c in registers, h -> d_out (f32) + hseq (bf16), device-scope flag barrier.
// ---------------------------------------------------------------------------
__global__ __launch_bounds__(256) void lstm_scan(
    const __bf16* __restrict__ whh_f,     // [4H, H] bf16
    const __bf16* __restrict__ whh_b,     // [4H, H] bf16
    const float* __restrict__ xg,         // [2, B*T, 4H]
    __bf16* __restrict__ hseq,            // [2, T, B, H] bf16
    float* __restrict__ out,              // [B, T, 2H]
    int* __restrict__ flags)              // [2, T], zeroed before launch
{
  extern __shared__ char smem[];
  __bf16* whh_s   = (__bf16*)smem;                      // [128][512]
  float*  gatebuf = (float*)(smem + 128 * HH * 2);      // [128][33] padded

  const int wg    = blockIdx.x;
  const int dir   = wg >> 4;
  const int slice = wg & 15;
  const int u0    = slice * 32;

  const __bf16* W    = dir ? whh_b : whh_f;
  const float*  xg_d = xg   + (size_t)dir * (size_t)(BB * TT) * G4;
  __bf16*       hs_d = hseq + (size_t)dir * (size_t)TT * BB * HH;
  int*          flg  = flags + dir * TT;

  const int tid = threadIdx.x;

  // Stage Whh slice: LDS row r -> gate (r>>5), unit u0 + (r&31).
  for (int idx = tid; idx < 128 * 64; idx += 256) {
    int r  = idx >> 6;
    int ch = idx & 63;           // 64 x uint4 per 512-elem row
    int grow = (r >> 5) * HH + u0 + (r & 31);
    *(uint4*)(whh_s + (size_t)r * HH + ch * 8) =
        *(const uint4*)(W + (size_t)grow * HH + ch * 8);
  }
  __syncthreads();

  const int wave  = tid >> 5;
  const int lane  = tid & 31;
  const int lm    = lane & 15;
  const int lhalf = lane >> 4;
  const int aoff  = lhalf * 8;
  const int boff  = lhalf * 8;

  // Wave w owns N-tile w: gate-columns w*16 .. w*16+15 of the 128-wide slice.
  const __bf16* bptr = whh_s + (size_t)(wave * 16 + lm) * HH;

  float creg[4];
  creg[0] = creg[1] = creg[2] = creg[3] = 0.0f;

  for (int t = 0; t < TT; ++t) {
    if (t > 0) {
      if (tid == 0) {
        while (__hip_atomic_load(&flg[t - 1], __ATOMIC_ACQUIRE,
                                 __HIP_MEMORY_SCOPE_AGENT) < 16) {
          __builtin_amdgcn_s_sleep(2);
        }
      }
      __syncthreads();
      __threadfence();
    }

    v8f acc0, acc1;
#pragma unroll
    for (int r = 0; r < 8; ++r) { acc0[r] = 0.0f; acc1[r] = 0.0f; }

    if (t > 0) {
      const __bf16* hprev = hs_d + (size_t)(t - 1) * BB * HH;
      const __bf16* ar0 = hprev + (size_t)lm * HH;          // batches 0..15
      const __bf16* ar1 = hprev + (size_t)(16 + lm) * HH;   // batches 16..31
      for (int kb = 0; kb < HH; kb += 32) {
        v16bf bf = ld_b_frag(bptr + kb + boff);
        v16bf a0 = ld_a_frag(ar0, kb + aoff);
        v16bf a1 = ld_a_frag(ar1, kb + aoff);
        acc0 = wmma_bf16(a0, bf, acc0);
        acc1 = wmma_bf16(a1, bf, acc1);
      }
    }

    // Exchange gate pre-activations through LDS (pad 33 avoids bank clash).
    {
      int col = wave * 16 + lm;
      float* gcol = gatebuf + col * 33;
#pragma unroll
      for (int r = 0; r < 8; ++r) {
        gcol[lhalf * 8 + r]      = acc0[r];
        gcol[16 + lhalf * 8 + r] = acc1[r];
      }
    }
    __syncthreads();

    // Elementwise LSTM cell: thread tid handles (b = k*8 + tid/32, j = tid&31).
    const int actual_t = dir ? (TT - 1 - t) : t;
#pragma unroll
    for (int k = 0; k < 4; ++k) {
      int p = k * 256 + tid;
      int b = p >> 5;
      int j = p & 31;
      const float* xgp = xg_d + (size_t)(b * TT + t) * G4 + u0 + j;
      float gi = gatebuf[(j)      * 33 + b] + xgp[0 * HH];
      float gf = gatebuf[(32 + j) * 33 + b] + xgp[1 * HH];
      float gg = gatebuf[(64 + j) * 33 + b] + xgp[2 * HH];
      float go = gatebuf[(96 + j) * 33 + b] + xgp[3 * HH];
      gi = sigmoidf_(gi);
      gf = sigmoidf_(gf);
      gg = tanhf(gg);
      go = sigmoidf_(go);
      float c = gf * creg[k] + gi * gg;
      creg[k] = c;
      float h = go * tanhf(c);
      out[(size_t)(b * TT + actual_t) * (2 * HH) + dir * HH + u0 + j] = h;
      hs_d[((size_t)t * BB + b) * HH + u0 + j] = (__bf16)h;
    }

    __threadfence();
    __syncthreads();
    if (tid == 0) {
      __hip_atomic_fetch_add(&flg[t], 1, __ATOMIC_RELEASE,
                             __HIP_MEMORY_SCOPE_AGENT);
    }
  }
}

// ---------------------------------------------------------------------------
// Host launcher
// ---------------------------------------------------------------------------
extern "C" void kernel_launch(void* const* d_in, const int* in_sizes, int n_in,
                              void* d_out, int out_size, void* d_ws, size_t ws_size,
                              hipStream_t stream) {
  const float* x     = (const float*)d_in[0];
  const float* Wih_f = (const float*)d_in[1];
  const float* Whh_f = (const float*)d_in[2];
  const float* bih_f = (const float*)d_in[3];
  const float* bhh_f = (const float*)d_in[4];
  const float* Wih_b = (const float*)d_in[5];
  const float* Whh_b = (const float*)d_in[6];
  const float* bih_b = (const float*)d_in[7];
  const float* bhh_b = (const float*)d_in[8];

  const size_t NW = (size_t)G4 * HH;        // 1,048,576 weight elems
  const size_t NX = (size_t)BB * TT * HH;   // 8,388,608 x elems

  char* ws = (char*)d_ws;
  size_t o = 0;
  int* flags = (int*)(ws + o);          o += 4096;
  __bf16* wihf = (__bf16*)(ws + o);     o += NW * 2;
  __bf16* whhf = (__bf16*)(ws + o);     o += NW * 2;
  __bf16* wihb = (__bf16*)(ws + o);     o += NW * 2;
  __bf16* whhb = (__bf16*)(ws + o);     o += NW * 2;
  __bf16* xbf  = (__bf16*)(ws + o);     o += NX * 2;
  __bf16* hseq = (__bf16*)(ws + o);     o += (size_t)2 * TT * BB * HH * 2;
  float*  xg   = (float*)(ws + o);      o += (size_t)2 * BB * TT * G4 * 4;

  hipMemsetAsync(flags, 0, 2 * TT * sizeof(int), stream);

  // bf16 conversions
  f32_to_bf16_kernel<<<(int)((NW + 255) / 256), 256, 0, stream>>>(Wih_f, wihf, (int)NW);
  f32_to_bf16_kernel<<<(int)((NW + 255) / 256), 256, 0, stream>>>(Whh_f, whhf, (int)NW);
  f32_to_bf16_kernel<<<(int)((NW + 255) / 256), 256, 0, stream>>>(Wih_b, wihb, (int)NW);
  f32_to_bf16_kernel<<<(int)((NW + 255) / 256), 256, 0, stream>>>(Whh_b, whhb, (int)NW);
  f32_to_bf16_kernel<<<(int)((NX + 255) / 256), 256, 0, stream>>>(x, xbf, (int)NX);

  // Phase 1: input GEMM for both directions
  dim3 g1(256, 8, 2);   // M=16384/64, N=2048/256, dir
  lstm_xgemm<<<g1, 256, 0, stream>>>(xbf, wihf, wihb,
                                     bih_f, bhh_f, bih_b, bhh_b, xg);

  // Phase 2: recurrent scan, 32 persistent WGs, 145KB dynamic LDS each
  size_t smem = (size_t)128 * HH * 2 + (size_t)128 * 33 * 4;
  hipFuncSetAttribute((const void*)lstm_scan,
                      hipFuncAttributeMaxDynamicSharedMemorySize, (int)smem);
  lstm_scan<<<32, 256, smem, stream>>>(whhf, whhb, xg, hseq,
                                       (float*)d_out, flags);
}
